// GRU_163208757461
// MI455X (gfx1250) — matrix operand order repaired and checked by reading, not compiled, not run on hardware
//
#include <hip/hip_runtime.h>
#include <hip/hip_bf16.h>

// ---------------------------------------------------------------------------
// GRU speech model on MI455X (gfx1250, wave32, WMMA, async-to-LDS DMA).
//
// Pipeline:
//   1) depthwise conv1d(K=3)+relu  -> bf16 activations
//   2) GRU layer 0: persistent kernel, 32 WGs; Wh slice pinned in LDS as
//      pre-swizzled bf16 WMMA B-tiles (192 KB); per-step the bf16 hidden
//      state (64 KB) and the step's x-slice (16 KB) are DMA'd into LDS with
//      global_load_async_to_lds_b128 (ASYNCcnt); fused x-projection with
//      B-tiles streamed from a pre-transposed bf16 W0x^T (L2-resident);
//      grid-wide spin barrier each of 1000 steps; fp32 master hidden state.
//   3) subsampled outputs -> xproj1 = ys0 @ W1x + b1[0]   (batched WMMA GEMM)
//   4) GRU layer 1: persistent kernel (precomputed xproj), 250 steps
//   5) dense: ys1 @ dense_w + dense_b (WMMA GEMM, N padded 41->48, masked)
// ---------------------------------------------------------------------------

typedef __bf16 bf16_t;
typedef __attribute__((ext_vector_type(16))) __bf16 v16bf;
typedef __attribute__((ext_vector_type(8)))  float  v8f;

union V16U { v16bf v; uint4 q[2]; };

#define B_      32
#define T0_     1000
#define F_      256
#define U_      1024
#define G3_     (3 * U_)
#define T1_     250
#define C_      41
#define NSLICE  32
#define GRU_THREADS 384

// LDS element budget for the GRU kernel (same for both layers):
//   Wh B-tiles: 32kt * 96cols * 32kk bf16   = 98304 elems (192 KB)
//   h stage   : 32  * 1024 bf16             = 32768 elems ( 64 KB)
//   x stage   : 32  * 256 bf16              =  8192 elems ( 16 KB)
//   gates     : 4 * 32*32 fp32                            ( 16 KB)
#define GRU_SMEM_BYTES ((98304 + 32768 + 8192) * 2 + 4 * B_ * 32 * 4)  // 294912

// A-matrix 16x32 bf16 per-lane pattern: 16 contiguous K values.
__device__ __forceinline__ v16bf load_a_bf16(const bf16_t* p) {
  V16U u;
  u.q[0] = *(const uint4*)(p);
  u.q[1] = *(const uint4*)(p + 8);
  return u.v;
}
// B-matrix 32x16 bf16 per-lane pattern: 16 contiguous K values.
__device__ __forceinline__ v16bf load_b_bf16(const bf16_t* p) {
  V16U u;
  u.q[0] = *(const uint4*)(p);
  u.q[1] = *(const uint4*)(p + 8);
  return u.v;
}
__device__ __forceinline__ v8f wmma_bf16(v16bf a, v16bf b, v8f c) {
  return __builtin_amdgcn_wmma_f32_16x16x32_bf16(
      /*neg_a=*/false, a, /*neg_b=*/false, b,
      /*c_mod=*/(short)0, c, /*reuse_a=*/false, /*reuse_b=*/false);
}
__device__ __forceinline__ float sigmoidf_(float x) {
  return 1.0f / (1.0f + __expf(-x));
}

// gfx1250 async DMA: copy 16 bytes global -> LDS per active lane.
// VDST VGPR holds the per-lane LDS byte offset, VADDR pair the 64-bit
// global address (GV mode). Tracked by ASYNCcnt (CDNA5 ISA).
__device__ __forceinline__ void async_copy_b128(unsigned int lds_off,
                                                const void* gaddr) {
  unsigned long long ga = (unsigned long long)(uintptr_t)gaddr;
  asm volatile("global_load_async_to_lds_b128 %0, %1, off"
               :: "v"(lds_off), "v"(ga)
               : "memory");
}
__device__ __forceinline__ void wait_async0() {
  asm volatile("s_wait_asynccnt 0x0" ::: "memory");
}

// ---------------------------------------------------------------------------
// 1) depthwise conv (kernel 3, SAME) + relu, fp32 -> bf16
// ---------------------------------------------------------------------------
__global__ void __launch_bounds__(256)
conv_relu_bf16(const float* __restrict__ x, const float* __restrict__ w,
               bf16_t* __restrict__ xc) {
  int i = blockIdx.x * blockDim.x + threadIdx.x;      // over B*T*F
  if (i >= B_ * T0_ * F_) return;
  int f  = i & (F_ - 1);
  int bt = i >> 8;                                     // F_ == 256
  int t  = bt % T0_;
  float s = x[i] * w[F_ + f];
  if (t > 0)        s += x[i - F_] * w[f];
  if (t < T0_ - 1)  s += x[i + F_] * w[2 * F_ + f];
  xc[i] = (bf16_t)fmaxf(s, 0.0f);
}

// ---------------------------------------------------------------------------
// weight transpose + convert: src fp32 [K][N] row-major -> dst bf16 [Nrows][K]
// (rows n >= Nvalid are zero padding)
// ---------------------------------------------------------------------------
__global__ void __launch_bounds__(256)
transpose_bf16(const float* __restrict__ src, bf16_t* __restrict__ dst,
               int K, int Nvalid, int Nrows) {
  int i = blockIdx.x * blockDim.x + threadIdx.x;      // over Nrows*K
  if (i >= Nrows * K) return;
  int n = i / K, k = i - n * K;
  dst[i] = (n < Nvalid) ? (bf16_t)src[(size_t)k * Nvalid + n] : (bf16_t)0.0f;
}

// ---------------------------------------------------------------------------
// hidden-state init: buffer 0 of the fp32 master copy + bf16 WMMA copy
// ---------------------------------------------------------------------------
__global__ void __launch_bounds__(256)
init_h(float* __restrict__ hf, bf16_t* __restrict__ hb,
       const float* __restrict__ h0 /* null -> zeros */) {
  int i = blockIdx.x * blockDim.x + threadIdx.x;      // over B*U
  if (i >= B_ * U_) return;
  float v = h0 ? h0[i & (U_ - 1)] : 0.0f;
  hf[i] = v;
  hb[i] = (bf16_t)v;
}

// ---------------------------------------------------------------------------
// batched GEMM: C[M x ldc] = A[M x K](bf16) * BT[N x K](bf16)^T + bias
// one 16x16 output tile per wave, K in 16x16x32 bf16 WMMA steps
// ---------------------------------------------------------------------------
__global__ void __launch_bounds__(256)
gemm_bf16_wmma(const bf16_t* __restrict__ A, const bf16_t* __restrict__ BT,
               const float* __restrict__ bias, float* __restrict__ C,
               int MT, int NT, int KT, int Nvalid, int ldc) {
  int wid  = (blockIdx.x * blockDim.x + threadIdx.x) >> 5;
  int lane = threadIdx.x & 31;
  int mt = wid / NT, nt = wid - mt * NT;
  if (mt >= MT) return;
  int K  = KT * 32;
  int n  = lane & 15;
  int hi = (lane >> 4) & 1;
  const bf16_t* arow = A  + (size_t)(mt * 16 + n) * K;
  const bf16_t* brow = BT + (size_t)(nt * 16 + n) * K;
  v8f acc = {};
  for (int kt = 0; kt < KT; ++kt) {
    v16bf a = load_a_bf16(arow + kt * 32 + hi * 16);
    v16bf b = load_b_bf16(brow + kt * 32 + hi * 16);
    acc = wmma_bf16(a, b, acc);
  }
  int col = nt * 16 + n;
  if (col < Nvalid) {
    float bv = bias ? bias[col] : 0.0f;
    for (int r = 0; r < 8; ++r) {
      int row = mt * 16 + r + hi * 8;
      C[(size_t)row * ldc + col] = acc[r] + bv;
    }
  }
}

// ---------------------------------------------------------------------------
// persistent GRU scan.
// grid = 32 WGs, each owns 32 U-columns (x 3 gates = 96 weight columns).
// LDS: Wh slice as pre-swizzled bf16 B-tiles [ktile][lc(96)][kk(32)] (192 KB)
//      + per-step async-staged h (64 KB) and x-slice (16 KB)
//      + 4 x 32x32 fp32 gate scratch.
// 12 waves: wave w -> (mt = w&1, nt = w>>1), nt -> gate g=nt>>1, half cg=nt&1.
// h double-buffered in global (fp32 master + bf16 matmul copy); per-step
// grid barrier via atomic counters.
// ---------------------------------------------------------------------------
__global__ void __launch_bounds__(GRU_THREADS)
gru_persistent(const float* __restrict__ Wh,    // [U][3U] fp32
               const bf16_t* __restrict__ BxT,  // [3U][Kx] bf16 (pre-transposed) or null
               const bf16_t* __restrict__ xin,  // [B][T][Kx] bf16 or null
               const float* __restrict__ xp,    // [B*T][3U] fp32 (precomputed, incl bias) or null
               const float* __restrict__ bx,    // [3U] or null
               const float* __restrict__ br,    // [3U]
               float* __restrict__ hf,          // [2][B][U] fp32
               bf16_t* __restrict__ hb,         // [2][B][U] bf16
               bf16_t* __restrict__ ysub,       // [B][T/sub][U] bf16
               int T, int sub, int ktx, int* __restrict__ counters) {
  extern __shared__ char smem[];
  const int KTH = U_ / 32;                        // 32 k-tiles for recurrence
  bf16_t* lwh   = (bf16_t*)smem;                  // 98304 elems
  bf16_t* lds_h = lwh + KTH * 96 * 32;            // 32768 elems
  bf16_t* lds_x = lds_h + B_ * U_;                // 8192 elems
  float*  lz    = (float*)(lds_x + B_ * F_);      // 32x32
  float*  lr    = lz  + B_ * 32;
  float*  lxh   = lr  + B_ * 32;
  float*  lrh   = lxh + B_ * 32;

  const int slice = blockIdx.x;                   // 0..31 -> U columns slice*32..+31
  const int tid   = threadIdx.x;
  const int kx    = ktx * 32;                     // x feature width (0 for layer 1)

  // ---- one-time: stage Wh slice into LDS (fp32 -> bf16, B-tile layout)
  for (int i = tid; i < U_ * 96; i += GRU_THREADS) {
    int k = i / 96, lc = i - (i / 96) * 96;
    int g = lc >> 5, cc = lc & 31;
    lwh[((k >> 5) * 96 + lc) * 32 + (k & 31)] =
        (bf16_t)Wh[(size_t)k * G3_ + g * U_ + slice * 32 + cc];
  }
  __syncthreads();

  const int wid  = tid >> 5;
  const int lane = tid & 31;
  const int mt   = wid & 1;        // batch 0-15 / 16-31
  const int nt   = wid >> 1;       // 0..5
  const int g    = nt >> 1;        // gate: 0=z 1=r 2=h
  const int cg   = nt & 1;         // 16-col half within the 32-col slice
  const int lcb  = g * 32 + cg * 16;
  const int n    = lane & 15;
  const int hi   = (lane >> 4) & 1;
  const int gcol = g * U_ + slice * 32 + cg * 16 + n;  // global weight column
  const int mrow = mt * 16 + n;                        // A row (batch index)

  const float bxv = bx ? bx[gcol] : 0.0f;
  const float brv = br[gcol];

  // LDS byte offsets for the async DMA destinations (flat->LDS: low 32 bits)
  const unsigned int h_lds_base = (unsigned int)(uintptr_t)lds_h;
  const unsigned int x_lds_base = (unsigned int)(uintptr_t)lds_x;

  for (int t = 0; t < T; ++t) {
    const int rb = t & 1, wb = rb ^ 1;

    // ---- async DMA: stage h(t) (and x(t)) into LDS, 16 B per lane per issue
    {
      const bf16_t* hsrc = hb + (size_t)rb * B_ * U_;
      for (int c = tid; c < (B_ * U_) / 8; c += GRU_THREADS)   // 4096 chunks
        async_copy_b128(h_lds_base + c * 16, hsrc + c * 8);
      if (xin) {
        const int cpr = kx / 8;                                // chunks per row
        for (int c = tid; c < B_ * cpr; c += GRU_THREADS) {    // 1024 chunks
          int b = c / cpr, o = (c - b * cpr) * 8;
          async_copy_b128(x_lds_base + c * 16,
                          xin + ((size_t)b * T + t) * kx + o);
        }
      }
      wait_async0();
      __syncthreads();
    }

    // ---- x contribution (B-tiles streamed from L2-resident W0x^T)
    v8f ax = {};
    if (xin) {
      const bf16_t* xrow = lds_x + (size_t)mrow * kx;
      const bf16_t* brow = BxT + (size_t)gcol * kx;
      for (int kt = 0; kt < ktx; ++kt) {
        v16bf a = load_a_bf16(xrow + kt * 32 + hi * 16);
        v16bf b = load_b_bf16(brow + kt * 32 + hi * 16);
        ax = wmma_bf16(a, b, ax);
      }
    } else {
      for (int r = 0; r < 8; ++r) {
        int row = mt * 16 + r + hi * 8;
        ax[r] = xp[((size_t)row * T + t) * G3_ + gcol];
      }
    }

    // ---- recurrent contribution: h(LDS) @ Wh(LDS)
    const bf16_t* hrow = lds_h + (size_t)mrow * U_;
    v8f ar = {};
    for (int kt = 0; kt < KTH; ++kt) {
      v16bf a = load_a_bf16(hrow + kt * 32 + hi * 16);
      v16bf b = load_b_bf16(lwh + ((size_t)(kt * 96 + lcb + n)) * 32 + hi * 16);
      ar = wmma_bf16(a, b, ar);
    }

    // ---- scatter gate pre-activations to LDS (C/D layout)
    const int cc2 = cg * 16 + n;                       // 0..31 within slice
    if (g == 0) {
      for (int r = 0; r < 8; ++r) {
        int m = mt * 16 + r + hi * 8;
        lz[m * 32 + cc2] = ax[r] + ar[r] + bxv + brv;
      }
    } else if (g == 1) {
      for (int r = 0; r < 8; ++r) {
        int m = mt * 16 + r + hi * 8;
        lr[m * 32 + cc2] = ax[r] + ar[r] + bxv + brv;
      }
    } else {
      for (int r = 0; r < 8; ++r) {
        int m = mt * 16 + r + hi * 8;
        lxh[m * 32 + cc2] = ax[r] + bxv;   // x part (r gate applies to rec only)
        lrh[m * 32 + cc2] = ar[r] + brv;
      }
    }
    __syncthreads();

    // ---- gate math + state update (fp32 master)
    for (int i = tid; i < B_ * 32; i += GRU_THREADS) {
      int m = i >> 5, cc = i & 31;
      float z  = sigmoidf_(lz[i]);
      float rr = sigmoidf_(lr[i]);
      float hh = tanhf(lxh[i] + rr * lrh[i]);
      int u = slice * 32 + cc;
      float hp = hf[(size_t)rb * B_ * U_ + m * U_ + u];
      float hn = z * hp + (1.0f - z) * hh;
      hf[(size_t)wb * B_ * U_ + m * U_ + u] = hn;
      hb[(size_t)wb * B_ * U_ + m * U_ + u] = (bf16_t)hn;
      if ((t & (sub - 1)) == 0)
        ysub[((size_t)m * (T / sub) + (t / sub)) * U_ + u] = (bf16_t)hn;
    }
    __threadfence();
    __syncthreads();

    // ---- grid-wide split barrier (32 co-resident WGs)
    if (t + 1 < T) {
      if (tid == 0) {
        atomicAdd(&counters[t], 1);
        while (atomicAdd(&counters[t], 0) < (int)gridDim.x)
          __builtin_amdgcn_s_sleep(1);
      }
      __syncthreads();
    }
  }
}

// ---------------------------------------------------------------------------
// host launcher
// ---------------------------------------------------------------------------
static inline size_t align256(size_t x) { return (x + 255) & ~(size_t)255; }

extern "C" void kernel_launch(void* const* d_in, const int* in_sizes, int n_in,
                              void* d_out, int out_size, void* d_ws, size_t ws_size,
                              hipStream_t stream) {
  (void)in_sizes; (void)n_in; (void)out_size; (void)ws_size;
  const float* x      = (const float*)d_in[0];   // (32,1000,256)
  const float* conv_w = (const float*)d_in[1];   // (3,1,256)
  const float* h0     = (const float*)d_in[2];   // (1,1024)
  const float* w0x    = (const float*)d_in[3];   // (256,3072)
  const float* w0h    = (const float*)d_in[4];   // (1024,3072)
  const float* b0     = (const float*)d_in[5];   // (2,3072)
  const float* w1x    = (const float*)d_in[6];   // (1024,3072)
  const float* w1h    = (const float*)d_in[7];   // (1024,3072)
  const float* b1     = (const float*)d_in[8];   // (2,3072)
  const float* dw     = (const float*)d_in[9];   // (1024,41)
  const float* db     = (const float*)d_in[10];  // (41,)
  float* out = (float*)d_out;                    // (32,250,41) fp32

  char* ws = (char*)d_ws;
  size_t off = 0;
  auto alloc = [&](size_t bytes) -> char* {
    char* p = ws + off;
    off = align256(off + bytes);
    return p;
  };
  bf16_t* xc   = (bf16_t*)alloc((size_t)B_ * T0_ * F_ * 2);     // conv out
  bf16_t* ys0  = (bf16_t*)alloc((size_t)B_ * T1_ * U_ * 2);     // layer0 subsampled
  float*  xp1  = (float*) alloc((size_t)B_ * T1_ * G3_ * 4);    // layer1 xproj
  bf16_t* ys1  = (bf16_t*)alloc((size_t)B_ * T1_ * U_ * 2);     // layer1 out
  bf16_t* w0xT = (bf16_t*)alloc((size_t)G3_ * F_ * 2);          // W0x^T bf16
  bf16_t* w1xT = (bf16_t*)alloc((size_t)G3_ * U_ * 2);          // W1x^T bf16
  bf16_t* dwT  = (bf16_t*)alloc((size_t)48 * U_ * 2);           // dense_w^T bf16 (pad 48)
  float*  hf   = (float*) alloc((size_t)2 * B_ * U_ * 4);       // h fp32 double-buffer
  bf16_t* hb   = (bf16_t*)alloc((size_t)2 * B_ * U_ * 2);       // h bf16 double-buffer
  int*    cnt  = (int*)   alloc((size_t)2048 * 4);              // barrier counters

  hipMemsetAsync(cnt, 0, 2048 * 4, stream);

  (void)hipFuncSetAttribute(reinterpret_cast<const void*>(gru_persistent),
                            hipFuncAttributeMaxDynamicSharedMemorySize,
                            GRU_SMEM_BYTES);

  // conv + weight prep (independent, throughput kernels)
  conv_relu_bf16<<<(B_ * T0_ * F_ + 255) / 256, 256, 0, stream>>>(x, conv_w, xc);
  transpose_bf16<<<(G3_ * F_ + 255) / 256, 256, 0, stream>>>(w0x, w0xT, F_, G3_, G3_);
  transpose_bf16<<<(G3_ * U_ + 255) / 256, 256, 0, stream>>>(w1x, w1xT, U_, G3_, G3_);
  transpose_bf16<<<(48 * U_ + 255) / 256, 256, 0, stream>>>(dw, dwT, U_, C_, 48);

  // layer 0: fused xproj (Kx=256), 1000 steps, subsample-by-4 output
  init_h<<<(B_ * U_ + 255) / 256, 256, 0, stream>>>(hf, hb, h0);
  gru_persistent<<<NSLICE, GRU_THREADS, GRU_SMEM_BYTES, stream>>>(
      w0h, w0xT, xc, nullptr, b0, b0 + G3_, hf, hb, ys0,
      T0_, /*sub=*/4, /*ktx=*/F_ / 32, cnt);

  // layer 1 input projection: ys0 (8000x1024) @ W1x (1024x3072) + b1[0]
  gemm_bf16_wmma<<<12000, 256, 0, stream>>>(ys0, w1xT, b1, xp1,
                                            /*MT=*/500, /*NT=*/192, /*KT=*/32,
                                            /*Nvalid=*/G3_, /*ldc=*/G3_);

  // layer 1: precomputed xproj, 250 steps
  init_h<<<(B_ * U_ + 255) / 256, 256, 0, stream>>>(hf, hb, nullptr);
  gru_persistent<<<NSLICE, GRU_THREADS, GRU_SMEM_BYTES, stream>>>(
      w1h, nullptr, nullptr, xp1, nullptr, b1 + G3_, hf, hb, ys1,
      T1_, /*sub=*/1, /*ktx=*/0, cnt + 1024);

  // dense: ys1 (8000x1024) @ dense_w (1024x41) + b -> d_out
  gemm_bf16_wmma<<<188, 256, 0, stream>>>(ys1, dwT, db, out,
                                          /*MT=*/500, /*NT=*/3, /*KT=*/32,
                                          /*Nvalid=*/C_, /*ldc=*/C_);
}
